// nnue_63977832841234
// MI455X (gfx1250) — matrix operand order repaired and checked by reading, not compiled, not run on hardware
//
#include <hip/hip_runtime.h>

// ---------------------------------------------------------------------------
// NNUE forward on MI455X (gfx1250, wave32, WMMA).
// Feature transform via v_wmma_f32_16x16x32_bf16, f32 accumulate.
// Double-buffered LDS (one barrier/K-step), 2Mx4N per-wave tiling to cut
// LDS fragment traffic to 12 b128-loads per 8 WMMAs.
// ---------------------------------------------------------------------------

typedef __attribute__((ext_vector_type(16))) __bf16 v16bf;
typedef __attribute__((ext_vector_type(8)))  __bf16 v8bf;
typedef __attribute__((ext_vector_type(8)))  float  v8f;

#define FT_IN   41600
#define KH      256
#define B_ROWS  4096
#define M_BLK   64
#define K_STEP  32
#define K_ITERS (FT_IN / K_STEP)   // 1300
#define LDS_PITCH 40               // 32 bf16 data + 8 pad (80B row, 16B aligned)

__device__ __forceinline__ float lclamp_f(float x) {
    return 0.01f * fminf(x, 0.0f)
         + fminf(fmaxf(x, 0.0f), 1.0f)
         + 0.01f * (fmaxf(x, 1.0f) - 1.0f);
}

__device__ __forceinline__ v8bf cvt8(const float4 a, const float4 b) {
    v8bf r;
    r[0] = (__bf16)a.x; r[1] = (__bf16)a.y; r[2] = (__bf16)a.z; r[3] = (__bf16)a.w;
    r[4] = (__bf16)b.x; r[5] = (__bf16)b.y; r[6] = (__bf16)b.z; r[7] = (__bf16)b.w;
    return r;
}

// ---------------------------------------------------------------------------
// h0[b, fsel*256 + n] = lclamp(feat[b,:] @ ft_w[n,:] + ft_b[n])
// Block: 256 threads (8 waves). Block tile: 64 rows x 256 cols.
// Wave tiling: mg = wave&1 -> M tiles {2mg, 2mg+1}; ng = wave>>1 -> N tiles
// {4ng .. 4ng+3}. Each wave: 2x4 accumulator tiles, 8 WMMAs per K-step.
// ---------------------------------------------------------------------------
__global__ __launch_bounds__(256, 2)
void ft_wmma_kernel(const float* __restrict__ feat1,
                    const float* __restrict__ feat2,
                    const float* __restrict__ ft_w,
                    const float* __restrict__ ft_b,
                    float* __restrict__ h0)
{
    __shared__ __align__(16) __bf16 lds_a[2][M_BLK][LDS_PITCH];
    __shared__ __align__(16) __bf16 lds_b[2][KH][LDS_PITCH];

    const int tid  = threadIdx.x;
    const int wave = tid >> 5;
    const int lane = tid & 31;
    const int lm   = lane & 15;   // row/col within 16x16 tile
    const int lh   = lane >> 4;   // half-wave select

    const int fsel  = blockIdx.y;                    // 0 -> features1, 1 -> features2
    const float* __restrict__ feat = fsel ? feat2 : feat1;
    const int mrow0 = blockIdx.x * M_BLK;

    const int mg = wave & 1;      // M tile group (2 tiles)
    const int ng = wave >> 1;     // N tile group (4 tiles)

    // Staging: A tile 64x32 f32 (4 threads/row, 8 floats each),
    //          B tile 256x32 f32 (1 thread per ft_w row, 32 floats).
    const int a_row  = tid >> 2;           // 0..63
    const int a_koff = (tid & 3) * 8;      // 0,8,16,24
    const float* aptr = feat + (size_t)(mrow0 + a_row) * FT_IN + a_koff;
    const float* bptr = ft_w + (size_t)tid * FT_IN;

    v8f acc[2][4] = {};

    // ---- prologue: stage buffer 0 ----
    {
        float4 ga0 = *(const float4*)(aptr + 0);
        float4 ga1 = *(const float4*)(aptr + 4);
        float4 gb[8];
        #pragma unroll
        for (int i = 0; i < 8; ++i) gb[i] = *(const float4*)(bptr + i * 4);
        aptr += K_STEP;
        bptr += K_STEP;
        *(v8bf*)&lds_a[0][a_row][a_koff] = cvt8(ga0, ga1);
        #pragma unroll
        for (int i = 0; i < 4; ++i)
            *(v8bf*)&lds_b[0][tid][i * 8] = cvt8(gb[2 * i], gb[2 * i + 1]);
        __syncthreads();
    }

    // ---- per-buffer compute: 4 a-loads + 8 b-loads (b128) -> 8 WMMAs ----
    auto compute = [&](int cur) {
        v16bf af[2];
        #pragma unroll
        for (int mi = 0; mi < 2; ++mi) {
            // A 16x32 bf16 lane layout: lane<16: m=lane, K 0..7 & 16..23;
            //                           lane>=16: m=lane-16, K 8..15 & 24..31
            const int ar = (mg * 2 + mi) * 16 + lm;
            const v8bf alo = *(const v8bf*)&lds_a[cur][ar][lh * 8];
            const v8bf ahi = *(const v8bf*)&lds_a[cur][ar][16 + lh * 8];
            #pragma unroll
            for (int i = 0; i < 8; ++i) { af[mi][i] = alo[i]; af[mi][8 + i] = ahi[i]; }
        }
        #pragma unroll
        for (int ni = 0; ni < 4; ++ni) {
            // B 32x16 bf16 lane layout: lane<16: n=lane, K 0..7 & 16..23;
            //                           lane>=16: n=lane-16, K 8..15 & 24..31
            const int br = ng * 64 + ni * 16 + lm;
            const v8bf blo = *(const v8bf*)&lds_b[cur][br][lh * 8];
            const v8bf bhi = *(const v8bf*)&lds_b[cur][br][16 + lh * 8];
            v16bf bf;
            #pragma unroll
            for (int i = 0; i < 8; ++i) { bf[i] = blo[i]; bf[8 + i] = bhi[i]; }
            #pragma unroll
            for (int mi = 0; mi < 2; ++mi)
                acc[mi][ni] = __builtin_amdgcn_wmma_f32_16x16x32_bf16(
                    false, af[mi], false, bf, (short)0, acc[mi][ni], false, false);
        }
    };

    // ---- main loop: one barrier per K-step, double-buffered LDS ----
    for (int kt = 0; kt < K_ITERS - 1; ++kt) {
        const int cur = kt & 1;
        const int nxt = cur ^ 1;

        // global loads for K-slice kt+1 (land while WMMAs run)
        float4 ga0 = *(const float4*)(aptr + 0);
        float4 ga1 = *(const float4*)(aptr + 4);
        float4 gb[8];
        #pragma unroll
        for (int i = 0; i < 8; ++i) gb[i] = *(const float4*)(bptr + i * 4);
        aptr += K_STEP;
        bptr += K_STEP;
        __builtin_prefetch(aptr, 0, 0);   // global_prefetch_b8, 2 slices ahead

        compute(cur);

        // convert + stage into the other buffer
        *(v8bf*)&lds_a[nxt][a_row][a_koff] = cvt8(ga0, ga1);
        #pragma unroll
        for (int i = 0; i < 4; ++i)
            *(v8bf*)&lds_b[nxt][tid][i * 8] = cvt8(gb[2 * i], gb[2 * i + 1]);

        __syncthreads();
    }
    compute((K_ITERS - 1) & 1);

    // ---- epilogue: bias + lclamp, write h0 [4096 x 512] f32 ----
    // C/D layout: VGPR v -> lanes 0-15: (M=v, N=lane); lanes 16-31: (M=8+v, N=lane-16)
    #pragma unroll
    for (int mi = 0; mi < 2; ++mi) {
        #pragma unroll
        for (int ni = 0; ni < 4; ++ni) {
            const int nloc = (ng * 4 + ni) * 16 + lm;   // 0..255
            const float bias = ft_b[nloc];
            const int col = fsel * KH + nloc;           // 0..511
            #pragma unroll
            for (int v = 0; v < 8; ++v) {
                const int row = mrow0 + (mg * 2 + mi) * 16 + lh * 8 + v;
                h0[(size_t)row * (2 * KH) + col] = lclamp_f(acc[mi][ni][v] + bias);
            }
        }
    }
}

// ---------------------------------------------------------------------------
// Tail: h1 = lclamp(h0 @ w1^T + b1); h2 = lclamp(h1 @ w2^T + b2); out = h2 @ wo^T + bo
// One thread per batch row. Weight indexing is wave-uniform -> scalar loads.
// ---------------------------------------------------------------------------
__global__ __launch_bounds__(256)
void tail_kernel(const float* __restrict__ h0,
                 const float* __restrict__ w1, const float* __restrict__ b1,
                 const float* __restrict__ w2, const float* __restrict__ b2,
                 const float* __restrict__ wo, const float* __restrict__ bo,
                 float* __restrict__ out)
{
    const int row = blockIdx.x * blockDim.x + threadIdx.x;
    if (row >= B_ROWS) return;
    const float* __restrict__ hr = h0 + (size_t)row * (2 * KH);

    float h1v[32];
    #pragma unroll
    for (int n = 0; n < 32; ++n) h1v[n] = b1[n];

    for (int k = 0; k < 2 * KH; k += 4) {
        const float x0 = hr[k + 0], x1 = hr[k + 1], x2 = hr[k + 2], x3 = hr[k + 3];
        #pragma unroll
        for (int n = 0; n < 32; ++n) {
            float a = h1v[n];
            a = fmaf(w1[n * 512 + k + 0], x0, a);
            a = fmaf(w1[n * 512 + k + 1], x1, a);
            a = fmaf(w1[n * 512 + k + 2], x2, a);
            a = fmaf(w1[n * 512 + k + 3], x3, a);
            h1v[n] = a;
        }
    }
    #pragma unroll
    for (int n = 0; n < 32; ++n) h1v[n] = lclamp_f(h1v[n]);

    float h2v[32];
    #pragma unroll
    for (int n = 0; n < 32; ++n) {
        float s = b2[n];
        #pragma unroll
        for (int k = 0; k < 32; ++k) s = fmaf(w2[n * 32 + k], h1v[k], s);
        h2v[n] = lclamp_f(s);
    }

    float s = bo[0];
    #pragma unroll
    for (int k = 0; k < 32; ++k) s = fmaf(wo[k], h2v[k], s);
    out[row] = s;
}

// ---------------------------------------------------------------------------
extern "C" void kernel_launch(void* const* d_in, const int* in_sizes, int n_in,
                              void* d_out, int out_size, void* d_ws, size_t ws_size,
                              hipStream_t stream)
{
    const float* f1   = (const float*)d_in[0];
    const float* f2   = (const float*)d_in[1];
    const float* ft_w = (const float*)d_in[2];
    const float* ft_b = (const float*)d_in[3];
    const float* w1   = (const float*)d_in[4];
    const float* b1   = (const float*)d_in[5];
    const float* w2   = (const float*)d_in[6];
    const float* b2   = (const float*)d_in[7];
    const float* wo   = (const float*)d_in[8];
    const float* bo   = (const float*)d_in[9];

    float* out = (float*)d_out;
    float* h0  = (float*)d_ws;            // 4096 * 512 f32 = 8 MB scratch

    dim3 grid(B_ROWS / M_BLK, 2);         // 64 M-blocks x {features1, features2}
    ft_wmma_kernel<<<grid, 256, 0, stream>>>(f1, f2, ft_w, ft_b, h0);
    tail_kernel<<<B_ROWS / 256, 256, 0, stream>>>(h0, w1, b1, w2, b2, wo, bo, out);
}
